// Encoder_73478300500500
// MI455X (gfx1250) — matrix operand order repaired and checked by reading, not compiled, not run on hardware
//
#include <hip/hip_runtime.h>
#include <hip/hip_bf16.h>
#include <math.h>

#define B_TOT  32768
#define HID    1024
#define VOC    9
#define TSTEPS 20
#define EOS_ID 8
#define KFC    784
#define KFCP   800   // K padded to multiple of 32
#define ROWS   32    // batch rows per block (2 WMMA M-tiles -> 2x B reuse)

typedef __bf16 v16bf __attribute__((ext_vector_type(16)));
typedef float  v8f   __attribute__((ext_vector_type(8)));

// A-fragment (16-bit 16x32): per lane, K-chunks {kh..kh+7} and {kh+16..kh+23}
__device__ __forceinline__ v16bf ld_a16(const __bf16* p) {
    union { v16bf v; uint4 q[2]; } u;
    u.q[0] = *reinterpret_cast<const uint4*>(p);
    u.q[1] = *reinterpret_cast<const uint4*>(p + 16);
    return u.v;
}
// B-fragment (16-bit 32x16): per lane, K-chunks {kh..kh+7} and {kh+16..kh+23}
__device__ __forceinline__ v16bf ld_b16(const __bf16* p) {
    union { v16bf v; uint4 q[2]; } u;
    u.q[0] = *reinterpret_cast<const uint4*>(p);
    u.q[1] = *reinterpret_cast<const uint4*>(p + 16);
    return u.v;
}
__device__ __forceinline__ v8f wmma_bf16(v16bf a, v16bf b, v8f c) {
    return __builtin_amdgcn_wmma_f32_16x16x32_bf16(false, a, false, b,
                                                   (short)0, c, false, false);
}

// Pre-convert weights to bf16 in workspace; pad fc_w K 784 -> 800 with zeros.
__global__ void cvt_weights(const float* __restrict__ w_hh,
                            const float* __restrict__ fc_w,
                            __bf16* __restrict__ whh_bf,
                            __bf16* __restrict__ fcw_bf) {
    int i = blockIdx.x * blockDim.x + threadIdx.x;
    if (i < 3 * HID * HID) whh_bf[i] = (__bf16)w_hh[i];
    if (i < HID * KFCP) {
        int r = i / KFCP, c = i - r * KFCP;
        fcw_bf[i] = (c < KFC) ? (__bf16)fc_w[r * KFC + c] : (__bf16)0.f;
    }
}

__global__ __launch_bounds__(256, 1) void gru_decoder(
    const float* __restrict__ x, const float* __restrict__ tau_p,
    const __bf16* __restrict__ fcw_bf, const __bf16* __restrict__ whh_bf,
    const float* __restrict__ w_ih, const float* __restrict__ b_ih,
    const float* __restrict__ b_hh, const float* __restrict__ proj_w,
    const float* __restrict__ proj_b, const float* __restrict__ gum,
    float* __restrict__ out_msg, int* __restrict__ out_len)
{
    extern __shared__ char smem_raw[];
    __bf16* sh_bf = (__bf16*)smem_raw;                          // 2 x 32x1024 bf16 (double buffer)
    float*  sh_f  = (float*)(smem_raw + 2 * ROWS * HID * 2);    // 32x1024 f32
    int*    tok_s = (int*)(smem_raw + 2 * ROWS * HID * 2 + ROWS * HID * 4);
    int*    fin_s = tok_s + ROWS;
    int*    len_s = fin_s + ROWS;
    __bf16* sx    = sh_bf + ROWS * HID;  // alias: buffer 1, only during phase 0

    const int tid  = threadIdx.x;
    const int lane = tid & 31;
    const int wv   = tid >> 5;          // 8 waves
    const int b0   = blockIdx.x * ROWS;
    const float tau = tau_p[0];

    if (tid < ROWS) { tok_s[tid] = 0; fin_s[tid] = 0; len_s[tid] = TSTEPS; }

    // Stage x tile as bf16, K padded with zeros
    for (int i = tid; i < ROWS * KFCP; i += 256) {
        int m = i / KFCP, k = i - m * KFCP;
        sx[i] = (k < KFC) ? (__bf16)x[(size_t)(b0 + m) * KFC + k] : (__bf16)0.f;
    }
    __syncthreads();

    const int mrow  = lane & 15;
    const int khalf = lane >> 4;
    const int kha   = khalf * 8;   // A-fragment K offset
    const int khb   = khalf * 8;   // B-fragment K offset

    // ---- Phase 0: h0 = gelu(x @ fc_w^T); two M-tiles share each B fragment ----
    for (int nt = 0; nt < 8; ++nt) {
        const int n0   = wv * 128 + nt * 16;
        const int ncol = n0 + mrow;
        v8f acc0 = {}, acc1 = {};
        const __bf16* ap0 = sx + mrow * KFCP + kha;
        const __bf16* ap1 = sx + (16 + mrow) * KFCP + kha;
        const __bf16* bp  = fcw_bf + (size_t)ncol * KFCP + khb;
        for (int k0 = 0; k0 < KFCP; k0 += 32) {
            v16bf b  = ld_b16(bp + k0);
            v16bf a0 = ld_a16(ap0 + k0);
            v16bf a1 = ld_a16(ap1 + k0);
            acc0 = wmma_bf16(a0, b, acc0);
            acc1 = wmma_bf16(a1, b, acc1);
        }
#pragma unroll
        for (int mt = 0; mt < 2; ++mt) {
            v8f acc = mt ? acc1 : acc0;
#pragma unroll
            for (int j = 0; j < 8; ++j) {
                int m = mt * 16 + khalf * 8 + j;
                float v = acc[j];
                float g = 0.5f * v * (1.f + erff(v * 0.70710678118654752f));
                sh_f[m * HID + ncol]  = g;
                sh_bf[m * HID + ncol] = (__bf16)g;   // buffer 0
            }
        }
    }
    __syncthreads();

    // ---- 20 GRU steps, hidden state stays in LDS ----
    for (int t = 0; t < TSTEPS; ++t) {
        const int cur = (t & 1) * (ROWS * HID);
        const int nxt = ((t + 1) & 1) * (ROWS * HID);

        for (int nt = 0; nt < 8; ++nt) {
            const int n0   = wv * 128 + nt * 16;
            const int ncol = n0 + mrow;
            v8f ar0 = {}, az0 = {}, an0 = {};
            v8f ar1 = {}, az1 = {}, an1 = {};
            const __bf16* ap0 = sh_bf + cur + mrow * HID + kha;
            const __bf16* ap1 = sh_bf + cur + (16 + mrow) * HID + kha;
            const __bf16* bpr = whh_bf + (size_t)(ncol) * HID + khb;
            const __bf16* bpz = whh_bf + (size_t)(HID + ncol) * HID + khb;
            const __bf16* bpn = whh_bf + (size_t)(2 * HID + ncol) * HID + khb;
            for (int k0 = 0; k0 < HID; k0 += 32) {
                __builtin_prefetch(bpr + k0 + 256, 0, 1);
                v16bf a0 = ld_a16(ap0 + k0);
                v16bf a1 = ld_a16(ap1 + k0);
                v16bf br = ld_b16(bpr + k0);
                v16bf bz = ld_b16(bpz + k0);
                v16bf bn = ld_b16(bpn + k0);
                ar0 = wmma_bf16(a0, br, ar0);
                ar1 = wmma_bf16(a1, br, ar1);
                az0 = wmma_bf16(a0, bz, az0);
                az1 = wmma_bf16(a1, bz, az1);
                an0 = wmma_bf16(a0, bn, an0);
                an1 = wmma_bf16(a1, bn, an1);
            }
            // GRU elementwise epilogue; gi is a gather (tokens are one-hot)
            float bihr = b_ih[ncol], bihz = b_ih[HID + ncol], bihn = b_ih[2 * HID + ncol];
            float bhhr = b_hh[ncol], bhhz = b_hh[HID + ncol], bhhn = b_hh[2 * HID + ncol];
#pragma unroll
            for (int mt = 0; mt < 2; ++mt) {
                v8f ar = mt ? ar1 : ar0;
                v8f az = mt ? az1 : az0;
                v8f an = mt ? an1 : an0;
#pragma unroll
                for (int j = 0; j < 8; ++j) {
                    int m   = mt * 16 + khalf * 8 + j;
                    int tok = tok_s[m];
                    float gir = w_ih[ncol * VOC + tok] + bihr;
                    float giz = w_ih[(HID + ncol) * VOC + tok] + bihz;
                    float gin = w_ih[(2 * HID + ncol) * VOC + tok] + bihn;
                    float hr = ar[j] + bhhr, hz = az[j] + bhhz, hn = an[j] + bhhn;
                    float r  = 1.f / (1.f + expf(-(gir + hr)));
                    float z  = 1.f / (1.f + expf(-(giz + hz)));
                    float nn = tanhf(gin + r * hn);
                    float hp = sh_f[m * HID + ncol];
                    float hv = (1.f - z) * nn + z * hp;
                    sh_f[m * HID + ncol]        = hv;       // owner-exclusive
                    sh_bf[nxt + m * HID + ncol] = (__bf16)hv;
                }
            }
        }
        __syncthreads();

        // ---- logits (V=9), gumbel argmax, one-hot emit: wave wv owns rows 4wv..4wv+3
        for (int rm = 0; rm < 4; ++rm) {
            const int m = wv * 4 + rm;
            float acc[VOC];
#pragma unroll
            for (int v = 0; v < VOC; ++v) acc[v] = 0.f;
            for (int k = lane; k < HID; k += 32) {
                float hv = sh_f[m * HID + k];
#pragma unroll
                for (int v = 0; v < VOC; ++v) acc[v] += hv * proj_w[v * HID + k];
            }
#pragma unroll
            for (int v = 0; v < VOC; ++v) {
                float rsum = acc[v];
                for (int off = 16; off > 0; off >>= 1)
                    rsum += __shfl_xor(rsum, off, 32);
                acc[v] = rsum;
            }
            if (lane == 0) {
                const int gb = b0 + m;
                const float* gp = gum + (size_t)t * B_TOT * VOC + (size_t)gb * VOC;
                int best = 0; float bestv = -3.4e38f;
#pragma unroll
                for (int v = 0; v < VOC; ++v) {
                    float lg = (acc[v] + proj_b[v] + gp[v]) / tau;
                    if (lg > bestv) { bestv = lg; best = v; }
                }
                float* om = out_msg + (size_t)gb * TSTEPS * VOC + t * VOC;
#pragma unroll
                for (int v = 0; v < VOC; ++v) om[v] = (v == best) ? 1.f : 0.f;
                tok_s[m] = best;
                if (best == EOS_ID && !fin_s[m]) { fin_s[m] = 1; len_s[m] = t + 1; }
            }
        }
        __syncthreads();
    }

    if (tid < ROWS) out_len[b0 + tid] = len_s[tid];
}

extern "C" void kernel_launch(void* const* d_in, const int* in_sizes, int n_in,
                              void* d_out, int out_size, void* d_ws, size_t ws_size,
                              hipStream_t stream) {
    const float* x      = (const float*)d_in[0];
    const float* tau    = (const float*)d_in[1];
    const float* fc_w   = (const float*)d_in[2];
    const float* w_ih   = (const float*)d_in[3];
    const float* w_hh   = (const float*)d_in[4];
    const float* b_ih   = (const float*)d_in[5];
    const float* b_hh   = (const float*)d_in[6];
    const float* proj_w = (const float*)d_in[7];
    const float* proj_b = (const float*)d_in[8];
    const float* gum    = (const float*)d_in[9];

    float* out_msg = (float*)d_out;
    int*   out_len = (int*)((float*)d_out + (size_t)B_TOT * TSTEPS * VOC);

    __bf16* whh_bf = (__bf16*)d_ws;
    __bf16* fcw_bf = whh_bf + (size_t)3 * HID * HID;

    const int ncvt = 3 * HID * HID;
    cvt_weights<<<(ncvt + 255) / 256, 256, 0, stream>>>(w_hh, fc_w, whh_bf, fcw_bf);

    const size_t smem = (size_t)2 * ROWS * HID * 2   // bf16 h double buffer
                      + (size_t)ROWS * HID * 4       // f32 h
                      + (size_t)3 * ROWS * 4;        // tok/fin/len
    gru_decoder<<<B_TOT / ROWS, 256, smem, stream>>>(
        x, tau, fcw_bf, whh_bf, w_ih, b_ih, b_hh, proj_w, proj_b, gum,
        out_msg, out_len);
}
